// CrossEncoderLongformer_31396210934224
// MI455X (gfx1250) — compile-verified, matrix operand order checked by reading, not run on hardware
//
#include <hip/hip_runtime.h>
#include <hip/hip_bf16.h>
#include <math.h>

// ---------------- model dims ----------------
#define Hdim 768
#define NHh  12
#define DHd  64
#define NLl  6
#define FFf  3072
#define Ww   256
#define CMAXc 32
#define Ll   4096
#define Gg   64
#define SCALEq 0.125f   // 1/sqrt(64)

typedef __bf16 bf16;
typedef __attribute__((ext_vector_type(16))) __bf16 v16bf;
typedef __attribute__((ext_vector_type(8)))  float  v8f;

// -------- gfx1250 async global->LDS path (guarded; falls back to sync staging) --------
#if defined(__gfx1250__) && __has_builtin(__builtin_amdgcn_global_load_async_to_lds_b128) && __has_builtin(__builtin_amdgcn_s_wait_asynccnt)
#define HAVE_ASYNC_LDS 1
typedef int i4v __attribute__((vector_size(16)));
typedef i4v __attribute__((address_space(1))) ga_i4v;   // global (device) int4
typedef i4v __attribute__((address_space(3))) ls_i4v;   // LDS (shared) int4
__device__ inline ga_i4v* as_global(const void* p) {
  return (ga_i4v*)(unsigned long long)p;
}
__device__ inline ls_i4v* as_local(const void* p) {
  // LDS generic pointers carry the LDS byte offset in the low 32 bits
  return (ls_i4v*)(unsigned int)(unsigned long long)p;
}
#else
#define HAVE_ASYNC_LDS 0
#endif

// ---------------- helpers ----------------
__device__ inline float gelu_f(float x) {
  return 0.5f * x * (1.0f + erff(x * 0.70710678118654752f));
}

__device__ inline float block_reduce_sum256(float v, float* red) {
  int tid = threadIdx.x;
  red[tid] = v; __syncthreads();
  #pragma unroll
  for (int s = 128; s >= 1; s >>= 1) {
    if (tid < s) red[tid] += red[tid + s];
    __syncthreads();
  }
  float r = red[0]; __syncthreads();
  return r;
}

// A fragment (16x32 bf16, 16-bit layout per CDNA5 ISA 7.12.2):
// lane (0..15 / 16..31) -> M = lane&15 ; vgpr i<4: K=2i+kh, i>=4: K=16+2(i-4)+kh, kh=8*(lane>=16)
__device__ inline v16bf load_a_frag(const bf16* As, int lda, int m_base, int lane) {
  int m  = m_base + (lane & 15);
  int kh = (lane >> 4) * 8;
  const bf16* rp = As + (size_t)m * lda;
  v16bf f;
  #pragma unroll
  for (int i = 0; i < 4; ++i) {
    int k0 = 2 * i + kh;
    f[2*i]   = rp[k0];
    f[2*i+1] = rp[k0 + 1];
  }
  #pragma unroll
  for (int i = 0; i < 4; ++i) {
    int k0 = 16 + 2 * i + kh;
    f[8+2*i]   = rp[k0];
    f[8+2*i+1] = rp[k0 + 1];
  }
  return f;
}

// B fragment (32x16 bf16, K x N), mirrored K packing; Bs row-major [32][ldb]
__device__ inline v16bf load_b_frag(const bf16* Bs, int ldb, int n_base, int lane) {
  int n  = n_base + (lane & 15);
  int kh = (lane >> 4) * 8;
  v16bf f;
  #pragma unroll
  for (int i = 0; i < 4; ++i) {
    int k0 = 2 * i + kh;
    f[2*i]   = Bs[(size_t)k0       * ldb + n];
    f[2*i+1] = Bs[(size_t)(k0 + 1) * ldb + n];
  }
  #pragma unroll
  for (int i = 0; i < 4; ++i) {
    int k0 = 16 + 2 * i + kh;
    f[8+2*i]   = Bs[(size_t)k0       * ldb + n];
    f[8+2*i+1] = Bs[(size_t)(k0 + 1) * ldb + n];
  }
  return f;
}

// fp32-LDS variants (convert to bf16 while building the fragment)
__device__ inline v16bf load_a_frag_f32(const float* As, int lda, int m_base, int lane) {
  int m  = m_base + (lane & 15);
  int kh = (lane >> 4) * 8;
  const float* rp = As + (size_t)m * lda;
  v16bf f;
  #pragma unroll
  for (int i = 0; i < 4; ++i) {
    int k0 = 2 * i + kh;
    f[2*i]   = (bf16)rp[k0];
    f[2*i+1] = (bf16)rp[k0 + 1];
  }
  #pragma unroll
  for (int i = 0; i < 4; ++i) {
    int k0 = 16 + 2 * i + kh;
    f[8+2*i]   = (bf16)rp[k0];
    f[8+2*i+1] = (bf16)rp[k0 + 1];
  }
  return f;
}

__device__ inline v16bf load_b_frag_f32(const float* Bs, int ldb, int n_base, int lane) {
  int n  = n_base + (lane & 15);
  int kh = (lane >> 4) * 8;
  v16bf f;
  #pragma unroll
  for (int i = 0; i < 4; ++i) {
    int k0 = 2 * i + kh;
    f[2*i]   = (bf16)Bs[(size_t)k0       * ldb + n];
    f[2*i+1] = (bf16)Bs[(size_t)(k0 + 1) * ldb + n];
  }
  #pragma unroll
  for (int i = 0; i < 4; ++i) {
    int k0 = 16 + 2 * i + kh;
    f[8+2*i]   = (bf16)Bs[(size_t)k0       * ldb + n];
    f[8+2*i+1] = (bf16)Bs[(size_t)(k0 + 1) * ldb + n];
  }
  return f;
}

// ---------------- embedding + LN ----------------
__global__ __launch_bounds__(256) void k_embed_ln(
    const float* __restrict__ emb_tok, const float* __restrict__ emb_pos,
    const float* __restrict__ sg, const float* __restrict__ bg,
    const int* __restrict__ ids, float* __restrict__ x) {
  __shared__ float red[256];
  int row = blockIdx.x, tid = threadIdx.x;
  int tok = ids[row];
  float e[3];
  #pragma unroll
  for (int i = 0; i < 3; ++i) {
    int c = tid + i * 256;
    e[i] = emb_tok[(size_t)tok * Hdim + c] + emb_pos[(size_t)row * Hdim + c];
  }
  float mean = block_reduce_sum256(e[0] + e[1] + e[2], red) * (1.0f / Hdim);
  float vs = 0.f;
  #pragma unroll
  for (int i = 0; i < 3; ++i) { float d = e[i] - mean; vs += d * d; }
  float var = block_reduce_sum256(vs, red) * (1.0f / Hdim);
  float inv = rsqrtf(var + 1e-5f);
  #pragma unroll
  for (int i = 0; i < 3; ++i) {
    int c = tid + i * 256;
    x[(size_t)row * Hdim + c] = (e[i] - mean) * inv * sg[c] + bg[c];
  }
}

// residual add + LN, in place on x
__global__ __launch_bounds__(256) void k_add_ln(
    float* __restrict__ x, const float* __restrict__ y,
    const float* __restrict__ sg, const float* __restrict__ bg) {
  __shared__ float red[256];
  int row = blockIdx.x, tid = threadIdx.x;
  float e[3];
  #pragma unroll
  for (int i = 0; i < 3; ++i) {
    int c = tid + i * 256;
    e[i] = x[(size_t)row * Hdim + c] + y[(size_t)row * Hdim + c];
  }
  float mean = block_reduce_sum256(e[0] + e[1] + e[2], red) * (1.0f / Hdim);
  float vs = 0.f;
  #pragma unroll
  for (int i = 0; i < 3; ++i) { float d = e[i] - mean; vs += d * d; }
  float var = block_reduce_sum256(vs, red) * (1.0f / Hdim);
  float inv = rsqrtf(var + 1e-5f);
  #pragma unroll
  for (int i = 0; i < 3; ++i) {
    int c = tid + i * 256;
    x[(size_t)row * Hdim + c] = (e[i] - mean) * inv * sg[c] + bg[c];
  }
}

// ---------------- WMMA GEMM: C = act(A[f32 MxK] @ B[f32 KxN] + bias) ----------------
// fp32 tiles staged to LDS (async DMA on gfx1250), bf16 conversion at fragment build.
#define BM 128
#define BN 128
#define BK 32
#define LDA_S (BK + 4)   // 36 floats -> 144B row stride (16B aligned)
#define LDB_S (BN + 4)   // 132 floats -> 528B row stride (16B aligned)

template<int ACT>
__global__ __launch_bounds__(256) void k_gemm(
    const float* __restrict__ A, const float* __restrict__ B,
    const float* __restrict__ bias, float* __restrict__ C,
    int M, int N, int K) {
  __shared__ float Asf[BM][LDA_S];
  __shared__ float Bsf[BK][LDB_S];
  int tid = threadIdx.x;
  int wave = tid >> 5, lane = tid & 31;
  int m0 = blockIdx.y * BM, n0 = blockIdx.x * BN;
  int wm0 = (wave >> 2) * 64;   // 2 waves along M
  int wn0 = (wave & 3) * 32;    // 4 waves along N
  v8f zacc = {0.f,0.f,0.f,0.f,0.f,0.f,0.f,0.f};
  v8f acc[4][2];
  #pragma unroll
  for (int i = 0; i < 4; ++i)
    #pragma unroll
    for (int j = 0; j < 2; ++j) acc[i][j] = zacc;

  for (int k0 = 0; k0 < K; k0 += BK) {
    // ---- stage A: 128x32 fp32 (4 x 16B chunks per thread) ----
    #pragma unroll
    for (int r = 0; r < 4; ++r) {
      int idx = tid + r * 256;
      int row = idx >> 3;
      int c4  = (idx & 7) * 4;
      int gm  = m0 + row;
      const float* gp = A + (size_t)gm * K + k0 + c4;
      float* lp = &Asf[row][c4];
      if (gm < M) {
#if HAVE_ASYNC_LDS
        __builtin_amdgcn_global_load_async_to_lds_b128(as_global(gp), as_local(lp), 0, 0);
#else
        *(float4*)lp = *(const float4*)gp;
#endif
      } else {
        *(float4*)lp = make_float4(0.f, 0.f, 0.f, 0.f);
      }
    }
    // ---- stage B: 32x128 fp32 ----
    #pragma unroll
    for (int r = 0; r < 4; ++r) {
      int idx = tid + r * 256;
      int kk  = idx >> 5;
      int c4  = (idx & 31) * 4;
      int gn  = n0 + c4;
      const float* gp = B + (size_t)(k0 + kk) * N + gn;
      float* lp = &Bsf[kk][c4];
      if (gn < N) {
#if HAVE_ASYNC_LDS
        __builtin_amdgcn_global_load_async_to_lds_b128(as_global(gp), as_local(lp), 0, 0);
#else
        *(float4*)lp = *(const float4*)gp;
#endif
      } else {
        *(float4*)lp = make_float4(0.f, 0.f, 0.f, 0.f);
      }
    }
    if (k0 + BK < K) {
      __builtin_prefetch(A + (size_t)(m0 + (tid >> 1)) * K + k0 + BK, 0, 3);
      __builtin_prefetch(B + (size_t)(k0 + BK + (tid & 31)) * N + n0, 0, 3);
    }
#if HAVE_ASYNC_LDS
    __builtin_amdgcn_s_wait_asynccnt(0);
#endif
    __syncthreads();

    v16bf af[4];
    #pragma unroll
    for (int i = 0; i < 4; ++i)
      af[i] = load_a_frag_f32(&Asf[0][0], LDA_S, wm0 + i * 16, lane);
    v16bf bfr[2];
    #pragma unroll
    for (int j = 0; j < 2; ++j)
      bfr[j] = load_b_frag_f32(&Bsf[0][0], LDB_S, wn0 + j * 16, lane);

    #pragma unroll
    for (int i = 0; i < 4; ++i)
      #pragma unroll
      for (int j = 0; j < 2; ++j)
        acc[i][j] = __builtin_amdgcn_wmma_f32_16x16x32_bf16(
            false, af[i], false, bfr[j], (short)0, acc[i][j], false, false);
    __syncthreads();
  }

  // epilogue: C layout -> row = r + 8*(lane>=16), col = lane&15
  int lh = lane >> 4, nn = lane & 15;
  #pragma unroll
  for (int i = 0; i < 4; ++i) {
    #pragma unroll
    for (int j = 0; j < 2; ++j) {
      int gn = n0 + wn0 + j * 16 + nn;
      if (gn < N) {
        float bb = bias[gn];
        #pragma unroll
        for (int r = 0; r < 8; ++r) {
          int gm = m0 + wm0 + i * 16 + r + 8 * lh;
          if (gm < M) {
            float vv = acc[i][j][r] + bb;
            if (ACT == 1) vv = gelu_f(vv);
            C[(size_t)gm * N + gn] = vv;
          }
        }
      }
    }
  }
}

// ---------------- local (banded) attention, flash-style with WMMA ----------------
// grid: (NH, Nb=16), 256 threads. Keys per block: 24 local tiles of 32 + 2 global tiles.
__global__ __launch_bounds__(256) void k_attn_local(
    const float* __restrict__ q, const float* __restrict__ k, const float* __restrict__ v,
    const float* __restrict__ kgs, const float* __restrict__ vgs,   // [G][H]
    const float* __restrict__ keybias, float* __restrict__ out) {
  __shared__ bf16 Qs[256][DHd + 2];
  __shared__ bf16 Kt[DHd][32 + 2];      // transposed key tile [d][j]
  __shared__ bf16 Vs[32][DHd + 2];      // value tile [j][d]
  __shared__ bf16 Ps[8][16][32 + 2];    // per-wave probability tile

  int h  = blockIdx.x;
  int nb = blockIdx.y;
  int tid = threadIdx.x, wave = tid >> 5, lane = tid & 31;
  int qrow0 = nb * Ww;
  int col0  = h * DHd;

  for (int i = tid; i < 256 * DHd; i += 256) {
    int r = i >> 6, d = i & 63;
    Qs[r][d] = (bf16)q[(size_t)(qrow0 + r) * Hdim + col0 + d];
  }
  __syncthreads();

  v8f zacc = {0.f,0.f,0.f,0.f,0.f,0.f,0.f,0.f};
  v8f o[2][4];
  float mrow[2][8], lrow[2][8];
  #pragma unroll
  for (int si = 0; si < 2; ++si) {
    #pragma unroll
    for (int t = 0; t < 4; ++t) o[si][t] = zacc;
    #pragma unroll
    for (int r = 0; r < 8; ++r) { mrow[si][r] = -1e30f; lrow[si][r] = 0.f; }
  }

  int lh = lane >> 4, nn = lane & 15;

  for (int t = 0; t < 26; ++t) {
    bool isglob = (t >= 24);
    // stage key/value tile (32 keys x 64 dims)
    for (int i = tid; i < 32 * DHd; i += 256) {
      int j = i >> 6, d = i & 63;
      float kv = 0.f, vv = 0.f;
      if (!isglob) {
        int kp = qrow0 - Ww + t * 32 + j;
        if (kp >= 0 && kp < Ll) {
          kv = k[(size_t)kp * Hdim + col0 + d];
          vv = v[(size_t)kp * Hdim + col0 + d];
        }
      } else {
        int g = (t - 24) * 32 + j;
        kv = kgs[(size_t)g * Hdim + col0 + d];
        vv = vgs[(size_t)g * Hdim + col0 + d];
      }
      Kt[d][j] = (bf16)kv;
      Vs[j][d] = (bf16)vv;
    }
    __syncthreads();

    v16bf kb[2][2];
    #pragma unroll
    for (int c = 0; c < 2; ++c)
      #pragma unroll
      for (int j = 0; j < 2; ++j)
        kb[c][j] = load_b_frag(&Kt[c * 32][0], 32 + 2, j * 16, lane);
    v16bf vb[4];
    #pragma unroll
    for (int j = 0; j < 4; ++j)
      vb[j] = load_b_frag(&Vs[0][0], DHd + 2, j * 16, lane);

    for (int si = 0; si < 2; ++si) {
      int strip = wave + si * 8;
      v16bf a0 = load_a_frag(&Qs[strip * 16][0],  DHd + 2, 0, lane);
      v16bf a1 = load_a_frag(&Qs[strip * 16][32], DHd + 2, 0, lane);

      v8f s0 = zacc, s1 = zacc;
      s0 = __builtin_amdgcn_wmma_f32_16x16x32_bf16(false, a0, false, kb[0][0], (short)0, s0, false, false);
      s0 = __builtin_amdgcn_wmma_f32_16x16x32_bf16(false, a1, false, kb[1][0], (short)0, s0, false, false);
      s1 = __builtin_amdgcn_wmma_f32_16x16x32_bf16(false, a0, false, kb[0][1], (short)0, s1, false, false);
      s1 = __builtin_amdgcn_wmma_f32_16x16x32_bf16(false, a1, false, kb[1][1], (short)0, s1, false, false);

      float sv[2][8];
      #pragma unroll
      for (int j = 0; j < 2; ++j) {
        #pragma unroll
        for (int r = 0; r < 8; ++r) {
          float s = ((j == 0) ? s0[r] : s1[r]) * SCALEq;
          float bsc;
          if (!isglob) {
            int qi = strip * 16 + r + 8 * lh;
            int jf = t * 32 + j * 16 + nn;
            int kp = qrow0 - Ww + jf;
            int rel = jf - Ww - qi;
            bool ok = (kp >= 0) && (kp < Ll) && (rel >= -Ww) && (rel <= Ww);
            int kpc = min(max(kp, 0), Ll - 1);
            bsc = ok ? keybias[kpc] : -1e9f;
          } else bsc = 0.f;
          sv[j][r] = s + bsc;
        }
      }
      // online softmax update (per-row reductions within 16-lane groups)
      float newm[8];
      #pragma unroll
      for (int r = 0; r < 8; ++r) {
        float mx = fmaxf(sv[0][r], sv[1][r]);
        #pragma unroll
        for (int off = 8; off >= 1; off >>= 1) mx = fmaxf(mx, __shfl_xor(mx, off, 16));
        newm[r] = fmaxf(mrow[si][r], mx);
      }
      #pragma unroll
      for (int r = 0; r < 8; ++r) {
        float sc = __expf(mrow[si][r] - newm[r]);
        lrow[si][r] *= sc;
        #pragma unroll
        for (int tt = 0; tt < 4; ++tt) o[si][tt][r] *= sc;
        mrow[si][r] = newm[r];
      }
      #pragma unroll
      for (int r = 0; r < 8; ++r) {
        float p0 = __expf(sv[0][r] - newm[r]);
        float p1 = __expf(sv[1][r] - newm[r]);
        sv[0][r] = p0; sv[1][r] = p1;
        float su = p0 + p1;
        #pragma unroll
        for (int off = 8; off >= 1; off >>= 1) su += __shfl_xor(su, off, 16);
        lrow[si][r] += su;
      }
      // relayout P (C-layout -> A-layout) through LDS
      #pragma unroll
      for (int j = 0; j < 2; ++j)
        #pragma unroll
        for (int r = 0; r < 8; ++r)
          Ps[wave][r + 8 * lh][j * 16 + nn] = (bf16)sv[j][r];
      __syncthreads();
      v16bf pf = load_a_frag(&Ps[wave][0][0], 32 + 2, 0, lane);
      #pragma unroll
      for (int tt = 0; tt < 4; ++tt)
        o[si][tt] = __builtin_amdgcn_wmma_f32_16x16x32_bf16(
            false, pf, false, vb[tt], (short)0, o[si][tt], false, false);
    }
    __syncthreads();
  }

  // write normalized output
  for (int si = 0; si < 2; ++si) {
    int strip = wave + si * 8;
    #pragma unroll
    for (int tt = 0; tt < 4; ++tt) {
      #pragma unroll
      for (int r = 0; r < 8; ++r) {
        int qi = strip * 16 + r + 8 * lh;
        int d  = tt * 16 + nn;
        float den = fmaxf(lrow[si][r], 1e-20f);
        out[(size_t)(qrow0 + qi) * Hdim + col0 + d] = o[si][tt][r] / den;
      }
    }
  }
}

// ---------------- global attention (overwrites rows at gpos) ----------------
__global__ __launch_bounds__(256) void k_attn_global(
    const float* __restrict__ qgs,   // [G][H]
    const float* __restrict__ kg,    // [L][H]
    const float* __restrict__ vg,    // [L][H]
    const unsigned char* __restrict__ am,
    const int* __restrict__ gpos, float* __restrict__ out) {
  __shared__ float sc[Ll];
  __shared__ float red[256];
  __shared__ float qrow[DHd];
  int h = blockIdx.x, g = blockIdx.y, tid = threadIdx.x;
  int col0 = h * DHd;
  if (tid < DHd) qrow[tid] = qgs[(size_t)g * Hdim + col0 + tid];
  __syncthreads();
  for (int l = tid; l < Ll; l += 256) {
    float s = 0.f;
    #pragma unroll 8
    for (int d = 0; d < DHd; ++d) s += qrow[d] * kg[(size_t)l * Hdim + col0 + d];
    s *= SCALEq;
    if (!am[l]) s = -1e9f;
    sc[l] = s;
  }
  __syncthreads();
  float mx = -1e30f;
  for (int l = tid; l < Ll; l += 256) mx = fmaxf(mx, sc[l]);
  red[tid] = mx; __syncthreads();
  #pragma unroll
  for (int s = 128; s >= 1; s >>= 1) { if (tid < s) red[tid] = fmaxf(red[tid], red[tid + s]); __syncthreads(); }
  mx = red[0]; __syncthreads();
  float sum = 0.f;
  for (int l = tid; l < Ll; l += 256) { float p = __expf(sc[l] - mx); sc[l] = p; sum += p; }
  red[tid] = sum; __syncthreads();
  #pragma unroll
  for (int s = 128; s >= 1; s >>= 1) { if (tid < s) red[tid] += red[tid + s]; __syncthreads(); }
  float inv = 1.0f / red[0]; __syncthreads();
  int d = tid & 63, gi = tid >> 6;
  float acc = 0.f;
  for (int l = gi; l < Ll; l += 4) acc += sc[l] * vg[(size_t)l * Hdim + col0 + d];
  red[tid] = acc; __syncthreads();
  if (tid < 64) {
    float r = red[tid] + red[tid + 64] + red[tid + 128] + red[tid + 192];
    out[(size_t)gpos[g] * Hdim + col0 + d] = r * inv;
  }
}

// ---------------- small utility kernels ----------------
__global__ void k_gather_rows(float* __restrict__ dst, const float* __restrict__ src,
                              const int* __restrict__ idx, int rowlen) {
  int r = blockIdx.x;
  int s = idx[r];
  for (int c = threadIdx.x; c < rowlen; c += blockDim.x)
    dst[(size_t)r * rowlen + c] = src[(size_t)s * rowlen + c];
}

__global__ void k_keybias(const unsigned char* __restrict__ am, float* __restrict__ kb) {
  int l = blockIdx.x * blockDim.x + threadIdx.x;
  if (l < Ll) kb[l] = am[l] ? 0.0f : -1e9f;
}

__global__ void k_keybias_glob(const int* __restrict__ gpos, float* __restrict__ kb) {
  int g = threadIdx.x;
  if (g < Gg) kb[gpos[g]] = -1e9f;
}

// stable argsort of (mask?0:1) == candidates first, then non-candidates, both in order
__global__ void k_select_cand(const int* __restrict__ ids, const int* __restrict__ cand,
                              int* __restrict__ positions, float* __restrict__ valid_out) {
  if (threadIdx.x == 0 && blockIdx.x == 0) {
    int ct = cand[0];
    int cnt = 0;
    for (int l = 0; l < Ll && cnt < CMAXc; ++l) if (ids[l] == ct) positions[cnt++] = l;
    for (int l = 0; l < Ll && cnt < CMAXc; ++l) if (ids[l] != ct) positions[cnt++] = l;
    for (int i = 0; i < CMAXc; ++i)
      valid_out[i] = (ids[positions[i]] == ct) ? 1.0f : 0.0f;
  }
}

__global__ __launch_bounds__(128) void k_logits(
    const float* __restrict__ hbuf, const float* __restrict__ Wh2,
    const float* __restrict__ bh2, float* __restrict__ out) {
  __shared__ float red[128];
  int i = blockIdx.x, tid = threadIdx.x;
  float acc = 0.f;
  for (int c = tid; c < Hdim / 2; c += 128) acc += hbuf[(size_t)i * (Hdim / 2) + c] * Wh2[c];
  red[tid] = acc; __syncthreads();
  #pragma unroll
  for (int s = 64; s >= 1; s >>= 1) { if (tid < s) red[tid] += red[tid + s]; __syncthreads(); }
  if (tid == 0) out[i] = red[0] + bh2[0];
}

// ---------------- orchestration ----------------
extern "C" void kernel_launch(void* const* d_in, const int* in_sizes, int n_in,
                              void* d_out, int out_size, void* d_ws, size_t ws_size,
                              hipStream_t stream) {
  const float* emb_tok = (const float*)d_in[0];
  const float* emb_pos = (const float*)d_in[1];
  const float* lnes    = (const float*)d_in[2];
  const float* lneb    = (const float*)d_in[3];
  const float* Wq  = (const float*)d_in[4];   const float* bq  = (const float*)d_in[5];
  const float* Wk  = (const float*)d_in[6];   const float* bk  = (const float*)d_in[7];
  const float* Wv  = (const float*)d_in[8];   const float* bv  = (const float*)d_in[9];
  const float* Wqg = (const float*)d_in[10];  const float* bqg = (const float*)d_in[11];
  const float* Wkg = (const float*)d_in[12];  const float* bkg = (const float*)d_in[13];
  const float* Wvg = (const float*)d_in[14];  const float* bvg = (const float*)d_in[15];
  const float* Wo  = (const float*)d_in[16];  const float* bo  = (const float*)d_in[17];
  const float* ln1s = (const float*)d_in[18]; const float* ln1b = (const float*)d_in[19];
  const float* W1  = (const float*)d_in[20];  const float* b1  = (const float*)d_in[21];
  const float* W2  = (const float*)d_in[22];  const float* b2  = (const float*)d_in[23];
  const float* ln2s = (const float*)d_in[24]; const float* ln2b = (const float*)d_in[25];
  const float* Wh1 = (const float*)d_in[26];  const float* bh1 = (const float*)d_in[27];
  const float* Wh2 = (const float*)d_in[28];  const float* bh2 = (const float*)d_in[29];
  const int* input_ids = (const int*)d_in[30];
  const unsigned char* am = (const unsigned char*)d_in[31];
  const int* gpos = (const int*)d_in[32];
  const int* cand = (const int*)d_in[33];
  float* outp = (float*)d_out;

  // carve workspace
  float* ws = (float*)d_ws;
  const size_t LH = (size_t)Ll * Hdim;
  float* x    = ws; ws += LH;
  float* tmp  = ws; ws += LH;
  float* qb   = ws; ws += LH;
  float* kb   = ws; ws += LH;
  float* vb   = ws; ws += LH;
  float* kgb  = ws; ws += LH;
  float* vgb  = ws; ws += LH;
  float* attn = ws; ws += LH;
  float* ffh  = ws; ws += (size_t)Ll * FFf;
  float* xg   = ws; ws += (size_t)Gg * Hdim;
  float* qgs  = ws; ws += (size_t)Gg * Hdim;
  float* kgs  = ws; ws += (size_t)Gg * Hdim;
  float* vgs  = ws; ws += (size_t)Gg * Hdim;
  float* keyb = ws; ws += Ll;
  float* gbuf = ws; ws += (size_t)CMAXc * Hdim;
  float* hbuf = ws; ws += (size_t)CMAXc * (Hdim / 2);
  int*   positions = (int*)ws;

  dim3 blk(256);
  dim3 grid_full((Hdim + BN - 1) / BN, (Ll + BM - 1) / BM);      // 6 x 32
  dim3 grid_ff1((FFf + BN - 1) / BN, (Ll + BM - 1) / BM);        // 24 x 32
  dim3 grid_ff2((Hdim + BN - 1) / BN, (Ll + BM - 1) / BM);
  dim3 grid_qg((Hdim + BN - 1) / BN, (Gg + BM - 1) / BM);        // 6 x 1
  dim3 grid_head(((Hdim / 2) + BN - 1) / BN, (CMAXc + BM - 1) / BM); // 3 x 1

  k_embed_ln<<<Ll, blk, 0, stream>>>(emb_tok, emb_pos, lnes, lneb, input_ids, x);
  k_keybias<<<Ll / 256, blk, 0, stream>>>(am, keyb);
  k_keybias_glob<<<1, 64, 0, stream>>>(gpos, keyb);

  const size_t HH = (size_t)Hdim * Hdim;
  for (int l = 0; l < NLl; ++l) {
    const float* Wq_l  = Wq  + l * HH;  const float* bq_l  = bq  + l * Hdim;
    const float* Wk_l  = Wk  + l * HH;  const float* bk_l  = bk  + l * Hdim;
    const float* Wv_l  = Wv  + l * HH;  const float* bv_l  = bv  + l * Hdim;
    const float* Wqg_l = Wqg + l * HH;  const float* bqg_l = bqg + l * Hdim;
    const float* Wkg_l = Wkg + l * HH;  const float* bkg_l = bkg + l * Hdim;
    const float* Wvg_l = Wvg + l * HH;  const float* bvg_l = bvg + l * Hdim;
    const float* Wo_l  = Wo  + l * HH;  const float* bo_l  = bo  + l * Hdim;
    const float* W1_l  = W1  + l * (size_t)Hdim * FFf; const float* b1_l = b1 + l * FFf;
    const float* W2_l  = W2  + l * (size_t)FFf * Hdim; const float* b2_l = b2 + l * Hdim;

    k_gemm<0><<<grid_full, blk, 0, stream>>>(x, Wq_l,  bq_l,  qb,  Ll, Hdim, Hdim);
    k_gemm<0><<<grid_full, blk, 0, stream>>>(x, Wk_l,  bk_l,  kb,  Ll, Hdim, Hdim);
    k_gemm<0><<<grid_full, blk, 0, stream>>>(x, Wv_l,  bv_l,  vb,  Ll, Hdim, Hdim);
    k_gemm<0><<<grid_full, blk, 0, stream>>>(x, Wkg_l, bkg_l, kgb, Ll, Hdim, Hdim);
    k_gemm<0><<<grid_full, blk, 0, stream>>>(x, Wvg_l, bvg_l, vgb, Ll, Hdim, Hdim);

    k_gather_rows<<<Gg, blk, 0, stream>>>(xg, x, gpos, Hdim);
    k_gemm<0><<<grid_qg, blk, 0, stream>>>(xg, Wqg_l, bqg_l, qgs, Gg, Hdim, Hdim);
    k_gather_rows<<<Gg, blk, 0, stream>>>(kgs, kgb, gpos, Hdim);
    k_gather_rows<<<Gg, blk, 0, stream>>>(vgs, vgb, gpos, Hdim);

    k_attn_local<<<dim3(NHh, Ll / Ww), blk, 0, stream>>>(qb, kb, vb, kgs, vgs, keyb, attn);
    k_attn_global<<<dim3(NHh, Gg), blk, 0, stream>>>(qgs, kgb, vgb, am, gpos, attn);

    k_gemm<0><<<grid_full, blk, 0, stream>>>(attn, Wo_l, bo_l, tmp, Ll, Hdim, Hdim);
    k_add_ln<<<Ll, blk, 0, stream>>>(x, tmp, ln1s + l * Hdim, ln1b + l * Hdim);

    k_gemm<1><<<grid_ff1, blk, 0, stream>>>(x, W1_l, b1_l, ffh, Ll, FFf, Hdim);
    k_gemm<0><<<grid_ff2, blk, 0, stream>>>(ffh, W2_l, b2_l, tmp, Ll, Hdim, FFf);
    k_add_ln<<<Ll, blk, 0, stream>>>(x, tmp, ln2s + l * Hdim, ln2b + l * Hdim);
  }

  k_select_cand<<<1, 1, 0, stream>>>(input_ids, cand, positions, outp + CMAXc);
  k_gather_rows<<<CMAXc, blk, 0, stream>>>(gbuf, x, positions, Hdim);
  k_gemm<1><<<grid_head, blk, 0, stream>>>(gbuf, Wh1, bh1, hbuf, CMAXc, Hdim / 2, Hdim);
  k_logits<<<CMAXc, 128, 0, stream>>>(hbuf, Wh2, bh2, outp);
}
